// BambaMixer_64544768524433
// MI455X (gfx1250) — matrix operand order repaired and checked by reading, not compiled, not run on hardware
//
#include <hip/hip_runtime.h>
#include <hip/hip_bf16.h>

// ---------------------------------------------------------------------------
// Bamba-9B mamba mixer forward for MI455X (gfx1250, wave32, WMMA bf16).
// Every GEMM operand is bf16 at load time. Big-GEMM B panels (64K x 64N)
// move via the Tensor Data Mover, double-buffered in LDS; one tensorcnt wait
// and two barriers per 8 WMMAs.
// ---------------------------------------------------------------------------

typedef __attribute__((ext_vector_type(16))) __bf16 v16bf;
typedef __attribute__((ext_vector_type(8)))  __bf16 v8bf;
typedef __attribute__((ext_vector_type(4)))  __bf16 v4bf;
typedef __attribute__((ext_vector_type(8)))  float  v8f;
typedef __attribute__((ext_vector_type(4)))  float  v4f;
typedef __attribute__((ext_vector_type(4)))  unsigned int v4u;
typedef __attribute__((ext_vector_type(8)))  int    v8i;
typedef __attribute__((ext_vector_type(4)))  int    v4i;

#define HID   2048
#define II    4096
#define NH    64
#define PP    64
#define NN    128
#define KCONV 4
#define CHUNK 256
#define NC    8
#define SEQ   2048
#define CONVD 4352
#define PROJW 8512

// Identity k -> LDS column (contiguous k order) so an A fragment is one contiguous 32B load.
__device__ __forceinline__ int swzA(int k) {
  return k;
}

// A fragment from bf16 row-major [M][K]: two 16B loads per lane.
__device__ __forceinline__ v16bf loadA_bf16(const __bf16* __restrict__ src, int ld,
                                            int lane) {
  int r = lane & 15, kh = (lane >> 4) & 1;
  const __bf16* p = src + (size_t)r * ld + kh * 16;
  v8bf lo = *(const v8bf*)(p);
  v8bf hi = *(const v8bf*)(p + 8);
  v16bf f;
#pragma unroll
  for (int i = 0; i < 8; ++i) { f[i] = lo[i]; f[8 + i] = hi[i]; }
  return f;
}
// B fragment from bf16 [N rows][K] (transposed-B storage): one 32B load.
__device__ __forceinline__ v16bf loadBt_bf16(const __bf16* __restrict__ src, int ld,
                                             int lane) {
  int n = lane & 15, kh = (lane >> 4) & 1;
  return *(const v16bf*)(src + (size_t)n * ld + kh * 16);
}
// A fragment from swizzled LDS bf16 [16][32]: one aligned 32B load.
__device__ __forceinline__ v16bf fragA_lds16(const __bf16* __restrict__ base,
                                             int row16, int lane) {
  int r = lane & 15, kh = (lane >> 4) & 1;
  return *(const v16bf*)(base + (size_t)(row16 + r) * 32 + kh * 16);
}
// B fragment from LDS bf16 [N rows][32 k]: one aligned 32B load.
__device__ __forceinline__ v16bf fragBt_lds16(const __bf16* __restrict__ base,
                                              int row, int lane) {
  int n = lane & 15, kh = (lane >> 4) & 1;
  return *(const v16bf*)(base + (size_t)(row + n) * 32 + kh * 16);
}
// B fragment from LDS bf16 [N rows][64 k] panel at k-offset ko (0 or 32).
__device__ __forceinline__ v16bf fragBt_lds64(const __bf16* __restrict__ base,
                                              int row, int ko, int lane) {
  int n = lane & 15, kh = (lane >> 4) & 1;
  return *(const v16bf*)(base + (size_t)(row + n) * 64 + ko + kh * 16);
}

__device__ __forceinline__ v8f wmma_bf16(v16bf a, v16bf b, v8f c) {
  return __builtin_amdgcn_wmma_f32_16x16x32_bf16(false, a, false, b, (short)0, c,
                                                 false, false);
}
__device__ __forceinline__ void store_tile(float* __restrict__ C, int ldc, v8f acc,
                                           int lane) {
  int n  = lane & 15;
  int mo = (lane >> 4) << 3;
#pragma unroll
  for (int v = 0; v < 8; ++v) C[(size_t)(mo + v) * ldc + n] = acc[v];
}
__device__ __forceinline__ v8f zero8() {
  v8f z = {0.f, 0.f, 0.f, 0.f, 0.f, 0.f, 0.f, 0.f};
  return z;
}

// --- Tensor Data Mover 2D tile -> LDS (ISA ch.8 D#). dsz: 0=1B,1=2B,2=4B ----
__device__ __forceinline__ void tdm_load_2d(unsigned lds_off, const void* gaddr,
                                            unsigned dsz, unsigned dim0,
                                            unsigned dim1, unsigned tile0,
                                            unsigned tile1,
                                            unsigned long long stride0) {
  unsigned long long ga = (unsigned long long)gaddr;
  v4u g0;
  g0[0] = 1u;
  g0[1] = lds_off;
  g0[2] = (unsigned)(ga & 0xffffffffu);
  g0[3] = (unsigned)((ga >> 32) & 0x1ffffffu) | (2u << 30);
  v8i g1;
  g1[0] = (int)(dsz << 16);
  g1[1] = (int)((dim0 & 0xffffu) << 16);
  g1[2] = (int)(((dim0 >> 16) & 0xffffu) | ((dim1 & 0xffffu) << 16));
  g1[3] = (int)(((dim1 >> 16) & 0xffffu) | (tile0 << 16));
  g1[4] = (int)(tile1 & 0xffffu);
  g1[5] = (int)(unsigned)(stride0 & 0xffffffffu);
  g1[6] = (int)(unsigned)((stride0 >> 32) & 0xffffu);
  g1[7] = 0;
  v4i z4 = {0, 0, 0, 0};
#if __clang_major__ >= 23
  v8i z8 = {0, 0, 0, 0, 0, 0, 0, 0};
  __builtin_amdgcn_tensor_load_to_lds(g0, g1, z4, z4, z8, 0);
#else
  __builtin_amdgcn_tensor_load_to_lds(g0, g1, z4, z4, 0);
#endif
}

// ---------------------------------------------------------------------------
__global__ __launch_bounds__(256) void f32_to_bf16(const float* __restrict__ in,
                                                   __bf16* __restrict__ outp,
                                                   long long n) {
  long long i = ((long long)blockIdx.x * 256 + threadIdx.x) * 4;
  if (i + 3 < n) {
    v4f q = *(const v4f*)(in + i);
    v4bf o;
#pragma unroll
    for (int j = 0; j < 4; ++j) o[j] = (__bf16)q[j];
    *(v4bf*)(outp + i) = o;
  }
}

// ---------------------------------------------------------------------------
// Direct bf16 GEMM (no LDS): C[M,N](f32) = A[M,K] @ B[N,K]^T, batched.
// grid=(M/128, N/64, batch)
// ---------------------------------------------------------------------------
__global__ __launch_bounds__(256) void gemm_bf16nn(
    const __bf16* __restrict__ A, const __bf16* __restrict__ B,
    float* __restrict__ C, int M, int N, int K,
    long long sA, long long sB, long long sC) {
  int lane = threadIdx.x & 31;
  int wave = threadIdx.x >> 5;
  const __bf16* Ab = A + (size_t)blockIdx.z * (size_t)sA;
  const __bf16* Bb = B + (size_t)blockIdx.z * (size_t)sB;
  float* Cb = C + (size_t)blockIdx.z * (size_t)sC;
  int m0 = (blockIdx.x * 8 + wave) * 16;
  int n0 = blockIdx.y * 64;
  if (m0 >= M) return;
  v8f acc[4] = {zero8(), zero8(), zero8(), zero8()};
  for (int k0 = 0; k0 < K; k0 += 32) {
    v16bf a = loadA_bf16(Ab + (size_t)m0 * K + k0, K, lane);
    v16bf bt[4];
#pragma unroll
    for (int t = 0; t < 4; ++t)
      bt[t] = loadBt_bf16(Bb + (size_t)(n0 + t * 16) * K + k0, K, lane);
#pragma unroll
    for (int t = 0; t < 4; ++t) acc[t] = wmma_bf16(a, bt[t], acc[t]);
  }
#pragma unroll
  for (int t = 0; t < 4; ++t)
    store_tile(Cb + (size_t)m0 * N + n0 + t * 16, N, acc[t], lane);
}

// ---------------------------------------------------------------------------
// Big GEMM: all-bf16. B panel (64K x 64N bf16, 8KB) via TDM, double-buffered
// LDS; 8 WMMAs per tensor-wait/barrier pair. K%64==0.
// ---------------------------------------------------------------------------
__global__ __launch_bounds__(256) void gemm_wmma_tdm_bf16(
    const __bf16* __restrict__ A, const __bf16* __restrict__ Bw,
    float* __restrict__ C, int M, int N, int K) {
  __shared__ __attribute__((aligned(64))) __bf16 Bpan[2][64 * 64];  // LDS base 0
  int lane = threadIdx.x & 31;
  int wave = threadIdx.x >> 5;
  int m0 = (blockIdx.x * 8 + wave) * 16;
  int n0 = blockIdx.y * 64;
  v8f acc[4] = {zero8(), zero8(), zero8(), zero8()};

  if (wave == 0)
    tdm_load_2d(0, Bw + (size_t)n0 * K, /*dsz=*/1u, (unsigned)K, (unsigned)N,
                /*tile_k=*/64, /*tile_n=*/64, (unsigned long long)K);

  for (int k0 = 0; k0 < K; k0 += 64) {
    int buf = (k0 >> 6) & 1;
    if (wave == 0) __builtin_amdgcn_s_wait_tensorcnt(0);
    __syncthreads();  // panel `buf` ready for all waves
    if (wave == 0 && k0 + 64 < K)
      tdm_load_2d((unsigned)((buf ^ 1) * 64 * 64 * 2),
                  Bw + (size_t)n0 * K + k0 + 64, 1u, (unsigned)K, (unsigned)N,
                  64, 64, (unsigned long long)K);
    // both A fragments issued together so their loads share one wait
    v16bf aA = loadA_bf16(A + (size_t)m0 * K + k0, K, lane);
    v16bf aB = loadA_bf16(A + (size_t)m0 * K + k0 + 32, K, lane);
    v16bf bf_[4];
#pragma unroll
    for (int t = 0; t < 4; ++t) bf_[t] = fragBt_lds64(&Bpan[buf][0], t * 16, 0, lane);
#pragma unroll
    for (int t = 0; t < 4; ++t) acc[t] = wmma_bf16(aA, bf_[t], acc[t]);
#pragma unroll
    for (int t = 0; t < 4; ++t) bf_[t] = fragBt_lds64(&Bpan[buf][0], t * 16, 32, lane);
#pragma unroll
    for (int t = 0; t < 4; ++t) acc[t] = wmma_bf16(aB, bf_[t], acc[t]);
    __syncthreads();  // done reading `buf` before it is refilled
  }
#pragma unroll
  for (int t = 0; t < 4; ++t)
    store_tile(C + (size_t)m0 * N + n0 + t * 16, N, acc[t], lane);
}

// ---------------------------------------------------------------------------
// Causal conv + SiLU + split. Emits xs/Bm fp32 (needed in fp32 downstream)
// and Bmb/Cmb bf16 (GEMM operands); softplus(dt).
// ---------------------------------------------------------------------------
__global__ void conv_silu_split(const float* __restrict__ proj,
                                const float* __restrict__ conv_w,
                                const float* __restrict__ conv_b,
                                const float* __restrict__ dt_bias,
                                float* __restrict__ xs, float* __restrict__ Bm,
                                __bf16* __restrict__ Bmb, __bf16* __restrict__ Cmb,
                                float* __restrict__ dt) {
  int d = blockIdx.x * 64 + threadIdx.x;
  int s = blockIdx.y;
  if (d < CONVD) {
    float acc = conv_b[d];
#pragma unroll
    for (int k = 0; k < KCONV; ++k) {
      int sr = s - (KCONV - 1) + k;
      float u = (sr >= 0) ? proj[(size_t)sr * PROJW + II + d] : 0.f;
      acc += u * conv_w[d * KCONV + k];
    }
    float y = acc / (1.f + __expf(-acc));
    if (d < II) {
      xs[(size_t)s * II + d] = y;
    } else if (d < II + NN) {
      Bm[s * NN + (d - II)] = y;
      Bmb[s * NN + (d - II)] = (__bf16)y;
    } else {
      Cmb[s * NN + (d - II - NN)] = (__bf16)y;
    }
  } else if (d < CONVD + NH) {
    int h = d - CONVD;
    float xr = proj[(size_t)s * PROJW + II + CONVD + h] + dt_bias[h];
    dt[s * NH + h] = (xr > 20.f) ? xr : log1pf(__expf(xr));
  }
}

// ---------------------------------------------------------------------------
__global__ void cumsum_dA(const float* __restrict__ dt, const float* __restrict__ A_log,
                          float* __restrict__ Acum) {
  int tid = blockIdx.x * blockDim.x + threadIdx.x;
  if (tid >= NC * NH) return;
  int c = tid >> 6, h = tid & 63;
  float A = -__expf(A_log[h]);
  size_t base = (size_t)c * CHUNK * NH + h;
  float run = 0.f;
  for (int l = 0; l < CHUNK; ++l) {
    run += dt[base + (size_t)l * NH] * A;
    Acum[base + (size_t)l * NH] = run;
  }
}

// ---------------------------------------------------------------------------
// Intra-chunk Y: masked decay panel (A-swizzled) + transposed X panel in LDS.
// ---------------------------------------------------------------------------
__global__ __launch_bounds__(256) void ssd_intra(
    const float* __restrict__ CB, const float* __restrict__ Acum,
    const float* __restrict__ dt, const float* __restrict__ xs,
    float* __restrict__ Y) {
  int h = blockIdx.x, c = blockIdx.y;
  __shared__ __attribute__((aligned(64))) __bf16 Mp[CHUNK * 32];
  __shared__ __attribute__((aligned(64))) __bf16 Xp[PP * 32];
  int lane = threadIdx.x & 31, wave = threadIdx.x >> 5;
  const float* CBc = CB + (size_t)c * CHUNK * CHUNK;
  const float* Ac  = Acum + (size_t)c * CHUNK * NH + h;
  const float* dtc = dt + (size_t)c * CHUNK * NH + h;
  const float* Xc  = xs + (size_t)c * CHUNK * II + h * PP;
  v8f acc[2][4] = {{zero8(), zero8(), zero8(), zero8()},
                   {zero8(), zero8(), zero8(), zero8()}};
  int t = threadIdx.x;
  float At = Ac[(size_t)t * NH];
  int p  = threadIdx.x & 63;
  int jg = threadIdx.x >> 6;
  for (int s0 = 0; s0 < CHUNK; s0 += 32) {
#pragma unroll 4
    for (int j = 0; j < 32; ++j) {
      int s = s0 + j;
      float m = 0.f;
      if (t >= s) m = CBc[t * CHUNK + s] * __expf(At - Ac[(size_t)s * NH]) * dtc[(size_t)s * NH];
      Mp[t * 32 + swzA(j)] = (__bf16)m;
    }
#pragma unroll
    for (int jj = 0; jj < 8; ++jj) {
      int j = jg * 8 + jj;
      Xp[p * 32 + j] = (__bf16)Xc[(size_t)(s0 + j) * II + p];
    }
    __syncthreads();
    v16bf bx[4];
#pragma unroll
    for (int nt = 0; nt < 4; ++nt) bx[nt] = fragBt_lds16(Xp, nt * 16, lane);
#pragma unroll
    for (int mi = 0; mi < 2; ++mi) {
      v16bf a = fragA_lds16(Mp, (wave + mi * 8) * 16, lane);
#pragma unroll
      for (int nt = 0; nt < 4; ++nt) acc[mi][nt] = wmma_bf16(a, bx[nt], acc[mi][nt]);
    }
    __syncthreads();
  }
  int n = lane & 15, mo = (lane >> 4) << 3;
#pragma unroll
  for (int mi = 0; mi < 2; ++mi) {
    int m0 = (wave + mi * 8) * 16;
#pragma unroll
    for (int nt = 0; nt < 4; ++nt)
#pragma unroll
      for (int v = 0; v < 8; ++v) {
        int mm = m0 + mo + v;
        Y[(size_t)(c * CHUNK + mm) * II + h * PP + nt * 16 + n] = acc[mi][nt][v];
      }
  }
}

// ---------------------------------------------------------------------------
// Chunk end states: A (p,l) swizzled + B (n,l) transposed, both LDS bf16.
// ---------------------------------------------------------------------------
__global__ __launch_bounds__(256) void ssd_states(
    const float* __restrict__ xs, const float* __restrict__ Bm,
    const float* __restrict__ Acum, const float* __restrict__ dt,
    float* __restrict__ states) {
  int h = blockIdx.x, c = blockIdx.y;
  __shared__ __attribute__((aligned(64))) __bf16 Ap[PP * 32];
  __shared__ __attribute__((aligned(64))) __bf16 Bpan[NN * 32];
  int lane = threadIdx.x & 31, wave = threadIdx.x >> 5;
  const float* Ac  = Acum + (size_t)c * CHUNK * NH + h;
  const float* dtc = dt + (size_t)c * CHUNK * NH + h;
  const float* Xc  = xs + (size_t)c * CHUNK * II + h * PP;
  const float* Bc  = Bm + (size_t)c * CHUNK * NN;
  float Aend = Ac[(size_t)(CHUNK - 1) * NH];
  int mt = wave & 3;
  int nb = (wave >> 2) * 64;
  int p   = threadIdx.x & 63;
  int jg4 = threadIdx.x >> 6;
  int nn  = threadIdx.x & 127;
  int jg2 = threadIdx.x >> 7;
  v8f acc[4] = {zero8(), zero8(), zero8(), zero8()};
  for (int s0 = 0; s0 < CHUNK; s0 += 32) {
#pragma unroll
    for (int jj = 0; jj < 8; ++jj) {
      int j = jg4 * 8 + jj;
      int l = s0 + j;
      float dte = __expf(Aend - Ac[(size_t)l * NH]) * dtc[(size_t)l * NH];
      Ap[p * 32 + swzA(j)] = (__bf16)(Xc[(size_t)l * II + p] * dte);
    }
#pragma unroll
    for (int jj = 0; jj < 16; ++jj) {
      int j = jg2 * 16 + jj;
      Bpan[nn * 32 + j] = (__bf16)Bc[(size_t)(s0 + j) * NN + nn];
    }
    __syncthreads();
    v16bf a = fragA_lds16(Ap, mt * 16, lane);
#pragma unroll
    for (int nt = 0; nt < 4; ++nt) {
      v16bf b = fragBt_lds16(Bpan, nb + nt * 16, lane);
      acc[nt] = wmma_bf16(a, b, acc[nt]);
    }
    __syncthreads();
  }
  float* So = states + (size_t)(c * NH + h) * PP * NN;
  int n = lane & 15, mo = (lane >> 4) << 3;
#pragma unroll
  for (int nt = 0; nt < 4; ++nt)
#pragma unroll
    for (int v = 0; v < 8; ++v)
      So[(size_t)(mt * 16 + mo + v) * NN + nb + nt * 16 + n] = acc[nt][v];
}

// ---------------------------------------------------------------------------
// Inter-chunk recurrence; emits chunk-start states directly as bf16.
// ---------------------------------------------------------------------------
__global__ __launch_bounds__(256) void ssd_scan(
    const float* __restrict__ Acum, const float* __restrict__ states,
    __bf16* __restrict__ prevb) {
  int h = blockIdx.x;
  float carry[32];
#pragma unroll
  for (int i = 0; i < 32; ++i) carry[i] = 0.f;
  for (int c = 0; c < NC; ++c) {
    float dec = __expf(Acum[(size_t)c * CHUNK * NH + (size_t)(CHUNK - 1) * NH + h]);
    size_t base = (size_t)(c * NH + h) * (PP * NN) + threadIdx.x;
#pragma unroll
    for (int i = 0; i < 32; ++i) {
      size_t idx = base + (size_t)i * 256;
      float st = states[idx];
      prevb[idx] = (__bf16)carry[i];
      carry[i] = carry[i] * dec + st;
    }
  }
}

// ---------------------------------------------------------------------------
// Inter-chunk Y + D skip: pure bf16 WMMA operands (Cmb, prevb).
// ---------------------------------------------------------------------------
__global__ __launch_bounds__(256) void ssd_inter(
    const __bf16* __restrict__ Cmb, const __bf16* __restrict__ prevb,
    const float* __restrict__ Acum, const float* __restrict__ xs,
    const float* __restrict__ Dp, float* __restrict__ Y) {
  int h = blockIdx.x, c = blockIdx.y;
  int lane = threadIdx.x & 31, wave = threadIdx.x >> 5;
  const __bf16* Cc = Cmb + (size_t)c * CHUNK * NN;
  const __bf16* Pv = prevb + (size_t)(c * NH + h) * PP * NN;  // [p][n], ld=NN
  const float* Ac = Acum + (size_t)c * CHUNK * NH + h;
  const float* Xc = xs + (size_t)c * CHUNK * II + h * PP;
  float Dh = Dp[h];
  v8f acc[2][4] = {{zero8(), zero8(), zero8(), zero8()},
                   {zero8(), zero8(), zero8(), zero8()}};
  for (int k0 = 0; k0 < NN; k0 += 32) {
    v16bf a0 = loadA_bf16(Cc + (size_t)(wave * 16) * NN + k0, NN, lane);
    v16bf a1 = loadA_bf16(Cc + (size_t)((wave + 8) * 16) * NN + k0, NN, lane);
    v16bf bt[4];
#pragma unroll
    for (int nt = 0; nt < 4; ++nt)
      bt[nt] = loadBt_bf16(Pv + (size_t)(nt * 16) * NN + k0, NN, lane);
#pragma unroll
    for (int nt = 0; nt < 4; ++nt) {
      acc[0][nt] = wmma_bf16(a0, bt[nt], acc[0][nt]);
      acc[1][nt] = wmma_bf16(a1, bt[nt], acc[1][nt]);
    }
  }
  int n = lane & 15, mo = (lane >> 4) << 3;
#pragma unroll
  for (int mi = 0; mi < 2; ++mi) {
    int m0 = (wave + mi * 8) * 16;
#pragma unroll
    for (int nt = 0; nt < 4; ++nt)
#pragma unroll
      for (int v = 0; v < 8; ++v) {
        int mm = m0 + mo + v;
        int pp = nt * 16 + n;
        size_t yi = (size_t)(c * CHUNK + mm) * II + h * PP + pp;
        float expA = __expf(Ac[(size_t)mm * NH]);
        Y[yi] = Y[yi] + expA * acc[mi][nt][v] + Dh * Xc[(size_t)mm * II + pp];
      }
  }
}

// ---------------------------------------------------------------------------
__global__ __launch_bounds__(256) void gated_rmsnorm(
    const float* __restrict__ proj, const float* __restrict__ norm_w,
    const float* __restrict__ Y, __bf16* __restrict__ Yb) {
  int s = blockIdx.x;
  __shared__ float red[256];
  float v[16];
  float ss = 0.f;
#pragma unroll
  for (int i = 0; i < 16; ++i) {
    int idx = threadIdx.x + i * 256;
    float y = Y[(size_t)s * II + idx];
    float g = proj[(size_t)s * PROJW + idx];
    float sg = g / (1.f + __expf(-g));
    v[i] = y * sg;
    ss += v[i] * v[i];
  }
  red[threadIdx.x] = ss;
  __syncthreads();
  for (int o = 128; o > 0; o >>= 1) {
    if ((int)threadIdx.x < o) red[threadIdx.x] += red[threadIdx.x + o];
    __syncthreads();
  }
  float scale = rsqrtf(red[0] / (float)II + 1e-5f);
#pragma unroll
  for (int i = 0; i < 16; ++i) {
    int idx = threadIdx.x + i * 256;
    Yb[(size_t)s * II + idx] = (__bf16)(v[i] * scale * norm_w[idx]);
  }
}

// ---------------------------------------------------------------------------
extern "C" void kernel_launch(void* const* d_in, const int* in_sizes, int n_in,
                              void* d_out, int out_size, void* d_ws, size_t ws_size,
                              hipStream_t stream) {
  (void)in_sizes; (void)n_in; (void)out_size; (void)ws_size;
  const float* x       = (const float*)d_in[0];
  const float* W_in    = (const float*)d_in[1];
  const float* conv_w  = (const float*)d_in[2];
  const float* conv_b  = (const float*)d_in[3];
  const float* dt_bias = (const float*)d_in[4];
  const float* A_log   = (const float*)d_in[5];
  const float* Dp      = (const float*)d_in[6];
  const float* norm_w  = (const float*)d_in[7];
  const float* W_out   = (const float*)d_in[8];
  float* out = (float*)d_out;

  float* ws     = (float*)d_ws;
  float* proj   = ws;                                   // SEQ*PROJW
  float* xs     = proj   + (size_t)SEQ * PROJW;         // SEQ*II
  float* Bm     = xs     + (size_t)SEQ * II;            // SEQ*NN
  float* dt     = Bm     + (size_t)SEQ * NN;            // SEQ*NH
  float* Acum   = dt     + (size_t)SEQ * NH;            // SEQ*NH
  float* CB     = Acum   + (size_t)SEQ * NH;            // NC*CHUNK*CHUNK
  float* states = CB     + (size_t)NC * CHUNK * CHUNK;  // NC*NH*PP*NN
  float* prev   = states + (size_t)NC * NH * PP * NN;   // NC*NH*PP*NN
  float* Y      = prev   + (size_t)NC * NH * PP * NN;   // SEQ*II
  __bf16* Bmb   = (__bf16*)(Y + (size_t)SEQ * II);      // SEQ*NN bf16
  __bf16* Cmb   = Bmb + (size_t)SEQ * NN;               // SEQ*NN bf16

  // bf16 staging aliases over regions that are dead at their time of use
  // (single stream => strict ordering):
  __bf16* xb     = (__bf16*)xs;      // dead once conv writes xs (after GEMM1)
  __bf16* Wb_in  = (__bf16*)states;  // dead after GEMM1; states written later
  __bf16* Wb_out = (__bf16*)CB;      // converted after CB/states are consumed
  __bf16* prevb  = (__bf16*)prev;    // written k7, read k8
  __bf16* Yb     = (__bf16*)prev;    // written k9 (prevb dead), read k10

  dim3 blk(256);
  // 0) stage x and W_in as bf16
  f32_to_bf16<<<dim3((SEQ * HID) / 1024), blk, 0, stream>>>(x, xb, (long long)SEQ * HID);
  f32_to_bf16<<<dim3((PROJW * HID) / 1024), blk, 0, stream>>>(W_in, Wb_in,
                                                              (long long)PROJW * HID);
  // 1) proj = x @ W_in^T (all-bf16, TDM B panels)
  gemm_wmma_tdm_bf16<<<dim3(SEQ / 128, PROJW / 64), blk, 0, stream>>>(
      xb, Wb_in, proj, SEQ, PROJW, HID);
  // 2) causal conv + SiLU + split + softplus(dt)
  conv_silu_split<<<dim3((CONVD + NH) / 64, SEQ), dim3(64), 0, stream>>>(
      proj, conv_w, conv_b, dt_bias, xs, Bm, Bmb, Cmb, dt);
  // 3) Acum = cumsum(dt * A)
  cumsum_dA<<<dim3(2), blk, 0, stream>>>(dt, A_log, Acum);
  // 4) CB[c] = C_chunk @ B_chunk^T (batched over chunks, pure bf16)
  gemm_bf16nn<<<dim3(CHUNK / 128, CHUNK / 64, NC), blk, 0, stream>>>(
      Cmb, Bmb, CB, CHUNK, CHUNK, NN,
      (long long)CHUNK * NN, (long long)CHUNK * NN, (long long)CHUNK * CHUNK);
  // 5) intra-chunk Y
  ssd_intra<<<dim3(NH, NC), blk, 0, stream>>>(CB, Acum, dt, xs, Y);
  // 6) per-chunk end states
  ssd_states<<<dim3(NH, NC), blk, 0, stream>>>(xs, Bm, Acum, dt, states);
  // 7) inter-chunk recurrence -> bf16 states
  ssd_scan<<<dim3(NH), blk, 0, stream>>>(Acum, states, prevb);
  // 7b) stage W_out as bf16 (CB + states regions are now dead)
  f32_to_bf16<<<dim3((HID * II) / 1024), blk, 0, stream>>>(W_out, Wb_out,
                                                           (long long)HID * II);
  // 8) inter-chunk Y + D skip (pure bf16 operands)
  ssd_inter<<<dim3(NH, NC), blk, 0, stream>>>(Cmb, prevb, Acum, xs, Dp, Y);
  // 9) gated RMSNorm -> bf16 (prev region is now dead)
  gated_rmsnorm<<<dim3(SEQ), blk, 0, stream>>>(proj, norm_w, Y, Yb);
  // 10) out = y @ W_out^T (all-bf16, TDM B panels)
  gemm_wmma_tdm_bf16<<<dim3(SEQ / 128, HID / 64), blk, 0, stream>>>(
      Yb, Wb_out, out, SEQ, HID, II);
}
